// ModuleNet2d_86706799771687
// MI455X (gfx1250) — matrix operand order repaired and checked by reading, not compiled, not run on hardware
//
#include <hip/hip_runtime.h>
#include <hip/hip_bf16.h>
#include <math.h>

typedef __attribute__((ext_vector_type(16))) _Float16 v16h;
typedef __attribute__((ext_vector_type(8)))  _Float16 v8h;
typedef __attribute__((ext_vector_type(8)))  float    v8f;

// ---- problem constants (from setup_inputs) ----
constexpr int B_ = 2, C_ = 64, h_ = 96, w_ = 96;
constexpr int H_ = 384, W_ = 384;
constexpr int D_ = 256;        // hidden dim
constexpr int CIN = 266;       // grid channels
constexpr int KP  = 288;       // CIN padded to multiple of 32
constexpr int NT  = 32;        // pixels per workgroup (two WMMA N-tiles)
constexpr float CELL_H = 2.0f / H_ * h_;   // = 0.5 (scale 4 -> max(scale/4,1)=1)
constexpr float CELL_W = 2.0f / W_ * w_;   // = 0.5

// ---- workspace layout (bytes) ----
constexpr size_t WS_WA = 0;                                  // f16 [256][288]
constexpr size_t WS_WB = WS_WA + (size_t)D_ * KP * 2;        // f16 [256][256]
constexpr size_t WS_SC = WS_WB + (size_t)D_ * D_ * 2;        // f32 [B][3][h][w]

__device__ __forceinline__ float coord1(int i, int n) {
    return -1.0f + 1.0f / n + (2.0f / n) * (float)i;
}

__device__ __forceinline__ v16h load_a_frag(const _Float16* p) {
    // elements 0..7 : K = base+0..7 ; elements 8..15 : K = base+16..23
    v8h lo = *(const v8h*)p;
    v8h hi = *(const v8h*)(p + 16);
    return __builtin_shufflevector(lo, hi, 0,1,2,3,4,5,6,7,8,9,10,11,12,13,14,15);
}

// ---------- prep: convert/pad weights to f16 ----------
__global__ void prep_weights(const float* __restrict__ w00, const float* __restrict__ w1,
                             _Float16* __restrict__ Wa, _Float16* __restrict__ Wb) {
    int i = blockIdx.x * blockDim.x + threadIdx.x;
    int n1 = D_ * KP;
    if (i < n1) {
        int o = i / KP, k = i % KP;
        Wa[i] = (_Float16)(k < CIN ? w00[o * CIN + k] : 0.0f);
    } else {
        int j = i - n1;
        if (j < D_ * D_) Wb[j] = (_Float16)w1[j];
    }
}

// ---------- shortcut branch: sc = ws2 @ relu(ws1 @ feat + bs1) + bs2 ----------
__global__ void shortcut_kernel(const float* __restrict__ feat,
                                const float* __restrict__ ws1, const float* __restrict__ bs1,
                                const float* __restrict__ ws2, const float* __restrict__ bs2,
                                float* __restrict__ sc) {
    int i = blockIdx.x * blockDim.x + threadIdx.x;
    if (i >= B_ * h_ * w_) return;
    int b   = i / (h_ * w_);
    int rem = i % (h_ * w_);
    const float* fp = feat + (size_t)b * C_ * h_ * w_ + rem;
    float f[C_];
#pragma unroll
    for (int c = 0; c < C_; ++c) f[c] = fp[c * h_ * w_];
    float o0 = bs2[0], o1 = bs2[1], o2 = bs2[2];
    for (int j = 0; j < C_; ++j) {
        float a = bs1[j];
        const float* wr = ws1 + j * C_;
#pragma unroll
        for (int c = 0; c < C_; ++c) a = fmaf(wr[c], f[c], a);
        a = fmaxf(a, 0.0f);
        o0 = fmaf(ws2[0 * C_ + j], a, o0);
        o1 = fmaf(ws2[1 * C_ + j], a, o1);
        o2 = fmaf(ws2[2 * C_ + j], a, o2);
    }
    sc[((size_t)b * 3 + 0) * h_ * w_ + rem] = o0;
    sc[((size_t)b * 3 + 1) * h_ * w_ + rem] = o1;
    sc[((size_t)b * 3 + 2) * h_ * w_ + rem] = o2;
}

// ---------- fused main kernel: 32 HR pixels per workgroup, 8 waves ----------
// Each wave: 32(M) x 32(N) output block = 2x2 WMMA tiles, A reused across N,
// B reused across M (4 wmma per 4 global-b128 + 4 ds-b128 per k-step).
__global__ __launch_bounds__(256)
void liif_main(const float* __restrict__ feat,
               const _Float16* __restrict__ Wa, const float* __restrict__ b00,
               const _Float16* __restrict__ Wb, const float* __restrict__ b1,
               const float* __restrict__ w2, const float* __restrict__ b2,
               const float* __restrict__ sc, float* __restrict__ out) {
    __shared__ __align__(32) _Float16 g_lds[NT * KP];   // grid, pixel-major [32][288]
    __shared__ __align__(32) _Float16 x_lds[NT * D_];   // layer0 out        [32][256]
    __shared__ __align__(32) _Float16 h_ldsm[NT * D_];  // gelu(layer1)      [32][256]
    __shared__ int   s_iy[4][NT], s_ix[4][NT];
    __shared__ float s_wq[4][NT];

    const int t    = threadIdx.x;
    const int pix0 = blockIdx.x * NT;
    const int b    = pix0 / (H_ * W_);
    const int rem  = pix0 % (H_ * W_);
    const int yy   = rem / W_;
    const int x0   = rem % W_;      // 32 | W, so row/batch uniform in WG

    // ---- phase 0: per-pixel geometry (threads 0..31) ----
    if (t < NT) {
        const int xx = x0 + t;
        const float cy = coord1(yy, H_);
        const float cx = coord1(xx, W_);
        const float rx = 1.0f / h_, ry = 1.0f / w_;
        float area[4];
#pragma unroll
        for (int qd = 0; qd < 4; ++qd) {
            float vx = (qd & 2) ? 1.0f : -1.0f;
            float vy = (qd & 1) ? 1.0f : -1.0f;
            float sy = fminf(fmaxf(cy + vx * rx + 1e-6f, -1.0f + 1e-6f), 1.0f - 1e-6f);
            float sx = fminf(fmaxf(cx + vy * ry + 1e-6f, -1.0f + 1e-6f), 1.0f - 1e-6f);
            int iy = (int)rintf(((sy + 1.0f) * h_ - 1.0f) * 0.5f);  // RNE == jnp.round
            int ix = (int)rintf(((sx + 1.0f) * w_ - 1.0f) * 0.5f);
            iy = min(max(iy, 0), h_ - 1);
            ix = min(max(ix, 0), w_ - 1);
            float rely = (cy - coord1(iy, h_)) * h_;
            float relx = (cx - coord1(ix, w_)) * w_;
            area[qd] = fabsf(rely * relx) + 1e-9f;
            s_iy[qd][t] = iy;
            s_ix[qd][t] = ix;
            g_lds[t * KP + 2 * qd + 0] = (_Float16)rely;
            g_lds[t * KP + 2 * qd + 1] = (_Float16)relx;
        }
        float tot = area[0] + area[1] + area[2] + area[3];
#pragma unroll
        for (int qd = 0; qd < 4; ++qd) s_wq[qd][t] = area[3 - qd] / tot;  // LIIF swap
        g_lds[t * KP + 264] = (_Float16)CELL_H;
        g_lds[t * KP + 265] = (_Float16)CELL_W;
#pragma unroll
        for (int k = CIN; k < KP; ++k) g_lds[t * KP + k] = (_Float16)0.0f;
    }
    __syncthreads();

    // ---- phase 1: weighted feature gather into grid ----
    {
        const int p = t & (NT - 1), s = t >> 5;   // 8 threads per pixel, 8 channels each
        const float* fb = feat + (size_t)b * C_ * h_ * w_;
#pragma unroll
        for (int qd = 0; qd < 4; ++qd) {
            int   idx = s_iy[qd][p] * w_ + s_ix[qd][p];
            float wq  = s_wq[qd][p];
#pragma unroll
            for (int u = 0; u < 8; ++u) {
                int c = s * 8 + u;
                g_lds[p * KP + 8 + qd * 64 + c] = (_Float16)(fb[c * h_ * w_ + idx] * wq);
            }
        }
    }
    __syncthreads();

    const int lane   = t & 31;
    const int wv     = t >> 5;
    const int p      = lane & 15;
    const int lh     = lane >> 4;       // lane half
    const int m_base = wv * 32;

    // ---- phase 2: GEMM1  x = Wa[256x288] @ grid[288x32] + b00 ----
    v8f a00 = {}, a01 = {}, a10 = {}, a11 = {};   // [mtile][ntile]
    for (int k0 = 0; k0 < KP; k0 += 32) {
        v16h A0 = load_a_frag(Wa + (size_t)(m_base + p) * KP + k0 + lh * 8);
        v16h A1 = load_a_frag(Wa + (size_t)(m_base + 16 + p) * KP + k0 + lh * 8);
        v16h B0 = load_a_frag(g_lds + p * KP + k0 + lh * 8);
        v16h B1 = load_a_frag(g_lds + (p + 16) * KP + k0 + lh * 8);
        a00 = __builtin_amdgcn_wmma_f32_16x16x32_f16(false, A0, false, B0, (short)0, a00, false, false);
        a01 = __builtin_amdgcn_wmma_f32_16x16x32_f16(false, A0, false, B1, (short)0, a01, false, false);
        a10 = __builtin_amdgcn_wmma_f32_16x16x32_f16(false, A1, false, B0, (short)0, a10, false, false);
        a11 = __builtin_amdgcn_wmma_f32_16x16x32_f16(false, A1, false, B1, (short)0, a11, false, false);
    }
#pragma unroll
    for (int r = 0; r < 8; ++r) {
        int m0 = m_base + r + lh * 8;
        int m1 = m0 + 16;
        float bm0 = b00[m0], bm1 = b00[m1];
        x_lds[p * D_ + m0]        = (_Float16)(a00[r] + bm0);
        x_lds[(p + 16) * D_ + m0] = (_Float16)(a01[r] + bm0);
        x_lds[p * D_ + m1]        = (_Float16)(a10[r] + bm1);
        x_lds[(p + 16) * D_ + m1] = (_Float16)(a11[r] + bm1);
    }
    __syncthreads();

    // ---- phase 3: GEMM2  h = gelu(Wb[256x256] @ x + b1) ----
    v8f c00 = {}, c01 = {}, c10 = {}, c11 = {};
    for (int k0 = 0; k0 < D_; k0 += 32) {
        v16h A0 = load_a_frag(Wb + (size_t)(m_base + p) * D_ + k0 + lh * 8);
        v16h A1 = load_a_frag(Wb + (size_t)(m_base + 16 + p) * D_ + k0 + lh * 8);
        v16h B0 = load_a_frag(x_lds + p * D_ + k0 + lh * 8);
        v16h B1 = load_a_frag(x_lds + (p + 16) * D_ + k0 + lh * 8);
        c00 = __builtin_amdgcn_wmma_f32_16x16x32_f16(false, A0, false, B0, (short)0, c00, false, false);
        c01 = __builtin_amdgcn_wmma_f32_16x16x32_f16(false, A0, false, B1, (short)0, c01, false, false);
        c10 = __builtin_amdgcn_wmma_f32_16x16x32_f16(false, A1, false, B0, (short)0, c10, false, false);
        c11 = __builtin_amdgcn_wmma_f32_16x16x32_f16(false, A1, false, B1, (short)0, c11, false, false);
    }
#pragma unroll
    for (int r = 0; r < 8; ++r) {
        int m0 = m_base + r + lh * 8;
        int m1 = m0 + 16;
        float v00 = c00[r] + b1[m0];
        float v01 = c01[r] + b1[m0];
        float v10 = c10[r] + b1[m1];
        float v11 = c11[r] + b1[m1];
        v00 = 0.5f * v00 * (1.0f + erff(v00 * 0.70710678118654752f));  // exact gelu
        v01 = 0.5f * v01 * (1.0f + erff(v01 * 0.70710678118654752f));
        v10 = 0.5f * v10 * (1.0f + erff(v10 * 0.70710678118654752f));
        v11 = 0.5f * v11 * (1.0f + erff(v11 * 0.70710678118654752f));
        h_ldsm[p * D_ + m0]        = (_Float16)v00;
        h_ldsm[(p + 16) * D_ + m0] = (_Float16)v01;
        h_ldsm[p * D_ + m1]        = (_Float16)v10;
        h_ldsm[(p + 16) * D_ + m1] = (_Float16)v11;
    }
    __syncthreads();

    // ---- phase 4: layer 256->3, shortcut bilinear sample, write out ----
    if (t < 3 * NT) {
        const int o  = t >> 5;          // 0..2
        const int pp = t & (NT - 1);    // 0..31
        const _Float16* hp = h_ldsm + pp * D_;
        const float*    wr = w2 + o * D_;
        float acc = b2[o];
        for (int k0 = 0; k0 < D_; k0 += 8) {
            v8h hv = *(const v8h*)(hp + k0);     // one ds_load_b128
#pragma unroll
            for (int u = 0; u < 8; ++u) acc = fmaf(wr[k0 + u], (float)hv[u], acc);
        }

        const int xx = x0 + pp;
        const float cy = coord1(yy, H_);
        const float cx = coord1(xx, W_);
        float uy = ((cy + 1.0f) * h_ - 1.0f) * 0.5f;
        float ux = ((cx + 1.0f) * w_ - 1.0f) * 0.5f;
        float fy = floorf(uy), fx = floorf(ux);
        float wy = uy - fy,    wx = ux - fx;
        int y0i = min(max((int)fy, 0), h_ - 1);
        int y1i = min(max((int)fy + 1, 0), h_ - 1);
        int x0i = min(max((int)fx, 0), w_ - 1);
        int x1i = min(max((int)fx + 1, 0), w_ - 1);
        const float* sp = sc + ((size_t)b * 3 + o) * h_ * w_;
        float samp = sp[y0i * w_ + x0i] * (1.0f - wy) * (1.0f - wx)
                   + sp[y0i * w_ + x1i] * (1.0f - wy) * wx
                   + sp[y1i * w_ + x0i] * wy * (1.0f - wx)
                   + sp[y1i * w_ + x1i] * wy * wx;
        out[((size_t)b * 3 + o) * H_ * W_ + (size_t)yy * W_ + xx] = acc + samp;
    }
}

extern "C" void kernel_launch(void* const* d_in, const int* in_sizes, int n_in,
                              void* d_out, int out_size, void* d_ws, size_t ws_size,
                              hipStream_t stream) {
    const float* feat = (const float*)d_in[0];
    const float* w00  = (const float*)d_in[1];
    const float* b00  = (const float*)d_in[2];
    const float* w1   = (const float*)d_in[3];
    const float* b1   = (const float*)d_in[4];
    const float* w2   = (const float*)d_in[5];
    const float* b2   = (const float*)d_in[6];
    const float* ws1  = (const float*)d_in[7];
    const float* bs1  = (const float*)d_in[8];
    const float* ws2  = (const float*)d_in[9];
    const float* bs2  = (const float*)d_in[10];
    float* out = (float*)d_out;

    char* ws = (char*)d_ws;
    _Float16* Wa = (_Float16*)(ws + WS_WA);
    _Float16* Wb = (_Float16*)(ws + WS_WB);
    float*    sc = (float*)(ws + WS_SC);

    int nprep = D_ * KP + D_ * D_;
    prep_weights<<<(nprep + 255) / 256, 256, 0, stream>>>(w00, w1, Wa, Wb);

    int nsc = B_ * h_ * w_;
    shortcut_kernel<<<(nsc + 255) / 256, 256, 0, stream>>>(feat, ws1, bs1, ws2, bs2, sc);

    int nblk = (B_ * H_ * W_) / NT;   // 9216 workgroups of 32 pixels
    liif_main<<<nblk, 256, 0, stream>>>(feat, Wa, b00, Wb, b1, w2, b2, sc, out);
}
